// TransformerMemorySystem_19524921328153
// MI455X (gfx1250) — matrix operand order repaired and checked by reading, not compiled, not run on hardware
//
#include <hip/hip_runtime.h>
#include <hip/hip_bf16.h>
#include <math.h>

// ---------------------------------------------------------------------------
// TransformerMemorySystem fused kernel for gfx1250 (MI455X), bf16 WMMA path.
//   B=16384, D=768, SLOTS=32
// Per 64-row tile (one workgroup, 8 wave32 waves, 252KB LDS):
//   stage cs (f32->bf16) + TDM-load mem_bf16 -> query GEMM -> scores
//   -> masked softmax(32) -> weighted -> gate GEMM (K=1536) -> sigmoid blend
// Weights + memory_buffer converted fp32->bf16 once into d_ws (~3.6MB, L2).
// ---------------------------------------------------------------------------

#define DIM   768
#define SLOTS 32
#define MTILE 64            // rows per workgroup
#define NT    (DIM / 16)    // 48 n-tiles
#define KT    (DIM / 32)    // 24 k-tiles (K=32 per bf16 wmma)
#define LDSM_OFF 196608u    // byte offset of LDS_M inside dynamic LDS

typedef __attribute__((ext_vector_type(16))) __bf16 bf16x16;
typedef __attribute__((ext_vector_type(8)))  float  f32x8;
typedef __attribute__((ext_vector_type(4)))  unsigned int u32x4;
typedef __attribute__((ext_vector_type(8)))  int    i32x8;
typedef __attribute__((ext_vector_type(4)))  int    i32x4;

union FragB16 { unsigned int u[8]; bf16x16 v; };

__device__ __forceinline__ unsigned short f2bf(float f) {
    unsigned int u = __builtin_bit_cast(unsigned int, f);
    unsigned int r = u + 0x7FFFu + ((u >> 16) & 1u);   // round-to-nearest-even
    return (unsigned short)(r >> 16);
}
__device__ __forceinline__ unsigned int pack2(float lo, float hi) {
    return (unsigned int)f2bf(lo) | ((unsigned int)f2bf(hi) << 16);
}
__device__ __forceinline__ float bf2f(unsigned short h) {
    unsigned int u = ((unsigned int)h) << 16;
    return __builtin_bit_cast(float, u);
}

// A fragment (16x32 bf16, MxK), src row-major [rows][stride] bf16.
__device__ __forceinline__ bf16x16 load_a(const unsigned short* src, int stride,
                                          int mbase, int kbase, int l16, int half) {
    FragB16 f;
    const unsigned short* row = src + (mbase + l16) * stride + kbase;
#pragma unroll
    for (int v = 0; v < 8; ++v) {
        int koff = half * 16 + v * 2;
        f.u[v] = *(const unsigned int*)(row + koff);
    }
    return f.v;
}

// B fragment (32x16 bf16, KxN) where memory holds W row-major as [N][ldk].
__device__ __forceinline__ bf16x16 load_bT(const unsigned short* w, int ldk,
                                           int nbase, int kbase, int l16, int half) {
    FragB16 f;
    const unsigned short* row = w + (size_t)(nbase + l16) * ldk + kbase + half * 16;
#pragma unroll
    for (int v = 0; v < 8; ++v)
        f.u[v] = *(const unsigned int*)(row + 2 * v);
    return f.v;
}

// B fragment where memory holds B row-major [K][ldn].
__device__ __forceinline__ bf16x16 load_b(const unsigned short* m, int ldn,
                                          int nbase, int kbase, int l16, int half) {
    FragB16 f;
    const unsigned short* col = m + nbase + l16;
#pragma unroll
    for (int v = 0; v < 8; ++v) {
        int k = kbase + 2 * v + 16 * half;
        unsigned int lo = col[(size_t)k * ldn];
        unsigned int hi = col[(size_t)(k + 1) * ldn];
        f.u[v] = lo | (hi << 16);
    }
    return f.v;
}

__global__ void cvt_bf16_kernel(const float* __restrict__ src,
                                unsigned short* __restrict__ dst, int n) {
    int i = blockIdx.x * 256 + threadIdx.x;
    if (i < n) dst[i] = f2bf(src[i]);
}

__global__ __launch_bounds__(256)
void tms_fused_kernel(const float* __restrict__ cs,          // [B,D]
                      const float* __restrict__ ctx,         // [B,D]
                      const unsigned char* __restrict__ mask,// [SLOTS]
                      const float* __restrict__ attn_b,      // [D]
                      const float* __restrict__ gate_b,      // [D]
                      const unsigned short* __restrict__ attn_w_bf, // [D][D]
                      const unsigned short* __restrict__ gate_w_bf, // [D][2D]
                      const unsigned short* __restrict__ memb_bf,   // [SLOTS][D]
                      float* __restrict__ out) {             // [B,D]
    extern __shared__ unsigned char smem[];
    unsigned short* LDS_A   = (unsigned short*)smem;               // 96KB cs, later weighted
    unsigned short* LDS_Q   = (unsigned short*)(smem + 98304);     // 96KB q, later ctx bf16
    unsigned short* LDS_M   = (unsigned short*)(smem + LDSM_OFF);  // 48KB mem bf16
    float*          LDS_S   = (float*)        (smem + 245760);     // 8KB scores f32
    unsigned short* LDS_ATT = (unsigned short*)(smem + 253952);    // 4KB attn bf16

    const int tid  = threadIdx.x;
    const int w    = tid >> 5;
    const int lane = tid & 31;
    const int l16  = lane & 15;
    const int half = lane >> 4;
    const int m0   = blockIdx.x * MTILE;

    // ---- Phase 0: stage cs tile (f32->bf16), TDM-load mem_bf16 -------------
    {
        const float4* src4 = (const float4*)(cs + (size_t)m0 * DIM);
        unsigned int* dst  = (unsigned int*)LDS_A;
        for (int i = tid; i < MTILE * DIM / 4; i += 256) {
            float4 x = src4[i];
            dst[i * 2]     = pack2(x.x, x.y);
            dst[i * 2 + 1] = pack2(x.z, x.w);
        }
    }
#if __has_builtin(__builtin_amdgcn_tensor_load_to_lds)
    if (w == 0) {
        // Tensor DMA descriptor (ISA 8.3/8.4): 2D bf16 tensor 768x32,
        // tile = whole tensor, dim0 stride 768 elements, dest LDS_M.
        unsigned long long ga = (unsigned long long)memb_bf;
        u32x4 g0 = { 1u,                                   // count=1, user D#
                     LDSM_OFF,                             // lds_addr (bytes)
                     (unsigned int)(ga & 0xffffffffu),
                     (unsigned int)(((ga >> 32) & 0x1ffffffu) | 0x80000000u) }; // type=2
        i32x8 g1 = { 0x00010000,      // data_size=1 (2 bytes), no flags
                     0x03000000,      // tensor_dim0[15:0]=768 @bits63:48
                     0x00200000,      // tensor_dim1[15:0]=32  @bits95:80
                     0x03000000,      // tile_dim0=768 @bits127:112
                     32,              // tile_dim1=32
                     768,             // tensor_dim0_stride lo
                     0, 0 };
        i32x4 gz4 = { 0, 0, 0, 0 };
        i32x8 gz8 = { 0, 0, 0, 0, 0, 0, 0, 0 };
        __builtin_amdgcn_tensor_load_to_lds(g0, g1, gz4, gz4, gz8, 0);
        __builtin_amdgcn_s_wait_tensorcnt(0);
    }
#else
    {
        const unsigned int* ms = (const unsigned int*)memb_bf;
        unsigned int* md = (unsigned int*)LDS_M;
        for (int i = tid; i < SLOTS * DIM / 2; i += 256) md[i] = ms[i];
    }
#endif
    __syncthreads();

    // ---- Phase 1: query = cs @ attn_w^T + attn_b  -> LDS_Q (bf16) ----------
    for (int j = 0; j < 3; ++j) {           // 2 n-tiles per pass: A reuse x2
        int nt0 = w + 8 * (2 * j);
        int nt1 = w + 8 * (2 * j + 1);
        f32x8 acc0[4], acc1[4];
#pragma unroll
        for (int mt = 0; mt < 4; ++mt) { acc0[mt] = (f32x8)0.0f; acc1[mt] = (f32x8)0.0f; }
        for (int kt = 0; kt < KT; ++kt) {
            if (kt + 1 < KT) {
                __builtin_prefetch(attn_w_bf + (size_t)(nt0 * 16 + l16) * DIM + (kt + 1) * 32, 0, 3);
                __builtin_prefetch(attn_w_bf + (size_t)(nt1 * 16 + l16) * DIM + (kt + 1) * 32, 0, 3);
            }
            bf16x16 b0 = load_bT(attn_w_bf, DIM, nt0 * 16, kt * 32, l16, half);
            bf16x16 b1 = load_bT(attn_w_bf, DIM, nt1 * 16, kt * 32, l16, half);
#pragma unroll
            for (int mt = 0; mt < 4; ++mt) {
                bf16x16 afr = load_a(LDS_A, DIM, mt * 16, kt * 32, l16, half);
                acc0[mt] = __builtin_amdgcn_wmma_f32_16x16x32_bf16(
                    false, afr, false, b0, (short)0, acc0[mt], false, false);
                acc1[mt] = __builtin_amdgcn_wmma_f32_16x16x32_bf16(
                    false, afr, false, b1, (short)0, acc1[mt], false, false);
            }
        }
        float bias0 = attn_b[nt0 * 16 + l16];
        float bias1 = attn_b[nt1 * 16 + l16];
#pragma unroll
        for (int mt = 0; mt < 4; ++mt)
#pragma unroll
            for (int r = 0; r < 8; ++r) {
                int m = mt * 16 + r + 8 * half;
                LDS_Q[m * DIM + nt0 * 16 + l16] = f2bf(acc0[mt][r] + bias0);
                LDS_Q[m * DIM + nt1 * 16 + l16] = f2bf(acc1[mt][r] + bias1);
            }
    }
    __syncthreads();

    // ---- Phase 2: scores = q @ mem^T  (64x32) -> LDS_S (f32) ---------------
    {
        int mt = w >> 1, nt = w & 1;        // 8 tiles, 1 per wave
        f32x8 acc = (f32x8)0.0f;
        for (int kt = 0; kt < KT; ++kt) {
            bf16x16 afr = load_a(LDS_Q, DIM, mt * 16, kt * 32, l16, half);
            bf16x16 bfr = load_bT(LDS_M, DIM, nt * 16, kt * 32, l16, half);
            acc = __builtin_amdgcn_wmma_f32_16x16x32_bf16(
                false, afr, false, bfr, (short)0, acc, false, false);
        }
        int n = nt * 16 + l16;
#pragma unroll
        for (int r = 0; r < 8; ++r)
            LDS_S[(mt * 16 + r + 8 * half) * SLOTS + n] = acc[r];
    }
    __syncthreads();

    // ---- Phase 3: masked softmax over 32 slots; also stage ctx bf16 --------
    if (tid < MTILE) {
        float s[SLOTS];
        float mx = -__builtin_inff();
#pragma unroll
        for (int j = 0; j < SLOTS; ++j) {
            float v = LDS_S[tid * SLOTS + j];
            v = mask[j] ? v : -__builtin_inff();
            s[j] = v;
            mx = fmaxf(mx, v);
        }
        float sum = 0.f;
#pragma unroll
        for (int j = 0; j < SLOTS; ++j) {
            float e = (s[j] == -__builtin_inff()) ? 0.f : __expf(s[j] - mx);
            s[j] = e; sum += e;
        }
        float inv = (sum > 0.f) ? (1.f / sum) : 0.f;
#pragma unroll
        for (int j = 0; j < SLOTS; ++j)
            LDS_ATT[tid * SLOTS + j] = f2bf(s[j] * inv);
    }
    {   // overwrite LDS_Q (q is dead) with bf16 memory_context tile
        const float4* src4 = (const float4*)(ctx + (size_t)m0 * DIM);
        unsigned int* dst  = (unsigned int*)LDS_Q;
        for (int i = tid; i < MTILE * DIM / 4; i += 256) {
            float4 x = src4[i];
            dst[i * 2]     = pack2(x.x, x.y);
            dst[i * 2 + 1] = pack2(x.z, x.w);
        }
    }
    __syncthreads();

    // ---- Phase 4: weighted = attn @ mem -> LDS_A (overwrites cs, bf16) -----
    for (int j = 0; j < 24; ++j) {          // 192 tiles / 8 waves
        int t  = w * 24 + j;
        int mt = t / NT, nt = t % NT;
        bf16x16 afr = load_a(LDS_ATT, SLOTS, mt * 16, 0, l16, half);
        bf16x16 bfr = load_b(LDS_M, DIM, nt * 16, 0, l16, half);
        f32x8 acc = (f32x8)0.0f;
        acc = __builtin_amdgcn_wmma_f32_16x16x32_bf16(
            false, afr, false, bfr, (short)0, acc, false, false);
        int n = nt * 16 + l16;
#pragma unroll
        for (int r = 0; r < 8; ++r)
            LDS_A[(mt * 16 + r + 8 * half) * DIM + n] = f2bf(acc[r]);
    }
    __syncthreads();

    // ---- Phase 5: gate GEMM (K=1536 over [weighted, ctx]) + sigmoid blend --
    for (int j = 0; j < 3; ++j) {
        int nt0 = w + 8 * (2 * j);
        int nt1 = w + 8 * (2 * j + 1);
        f32x8 acc0[4], acc1[4];
#pragma unroll
        for (int mt = 0; mt < 4; ++mt) { acc0[mt] = (f32x8)0.0f; acc1[mt] = (f32x8)0.0f; }
        for (int kt = 0; kt < 2 * KT; ++kt) {
            if (kt + 1 < 2 * KT) {
                __builtin_prefetch(gate_w_bf + (size_t)(nt0 * 16 + l16) * 2 * DIM + (kt + 1) * 32, 0, 3);
                __builtin_prefetch(gate_w_bf + (size_t)(nt1 * 16 + l16) * 2 * DIM + (kt + 1) * 32, 0, 3);
            }
            bf16x16 b0 = load_bT(gate_w_bf, 2 * DIM, nt0 * 16, kt * 32, l16, half);
            bf16x16 b1 = load_bT(gate_w_bf, 2 * DIM, nt1 * 16, kt * 32, l16, half);
            const unsigned short* asrc = (kt < KT) ? LDS_A : LDS_Q;
            int kb = (kt < KT) ? kt * 32 : (kt - KT) * 32;
#pragma unroll
            for (int mt = 0; mt < 4; ++mt) {
                bf16x16 afr = load_a(asrc, DIM, mt * 16, kb, l16, half);
                acc0[mt] = __builtin_amdgcn_wmma_f32_16x16x32_bf16(
                    false, afr, false, b0, (short)0, acc0[mt], false, false);
                acc1[mt] = __builtin_amdgcn_wmma_f32_16x16x32_bf16(
                    false, afr, false, b1, (short)0, acc1[mt], false, false);
            }
        }
#pragma unroll
        for (int p = 0; p < 2; ++p) {
            int n = (p == 0 ? nt0 : nt1) * 16 + l16;
            float gbias = gate_b[n];
#pragma unroll
            for (int mt = 0; mt < 4; ++mt)
#pragma unroll
                for (int r = 0; r < 8; ++r) {
                    int m = mt * 16 + r + 8 * half;
                    float a  = (p == 0 ? acc0[mt][r] : acc1[mt][r]);
                    float g  = 1.f / (1.f + __expf(-(a + gbias)));
                    float wv = bf2f(LDS_A[m * DIM + n]);
                    float cv = ctx[(size_t)(m0 + m) * DIM + n];
                    out[(size_t)(m0 + m) * DIM + n] = g * wv + (1.f - g) * cv;
                }
        }
    }
}

extern "C" void kernel_launch(void* const* d_in, const int* in_sizes, int n_in,
                              void* d_out, int out_size, void* d_ws, size_t ws_size,
                              hipStream_t stream) {
    const float*         cs     = (const float*)d_in[0];
    const float*         ctx    = (const float*)d_in[1];
    const float*         memb   = (const float*)d_in[2];
    const unsigned char* mask   = (const unsigned char*)d_in[3];
    const float*         attn_w = (const float*)d_in[4];
    const float*         attn_b = (const float*)d_in[5];
    const float*         gate_w = (const float*)d_in[6];
    const float*         gate_b = (const float*)d_in[7];
    float*               out    = (float*)d_out;

    const int Brows  = in_sizes[0] / DIM;          // 16384
    const int n_attn = DIM * DIM;                  // 589824
    const int n_gate = DIM * 2 * DIM;              // 1179648
    const int n_memb = SLOTS * DIM;                // 24576

    unsigned short* attn_w_bf = (unsigned short*)d_ws;
    unsigned short* gate_w_bf = attn_w_bf + n_attn;
    unsigned short* memb_bf   = gate_w_bf + n_gate;

    cvt_bf16_kernel<<<(n_attn + 255) / 256, 256, 0, stream>>>(attn_w, attn_w_bf, n_attn);
    cvt_bf16_kernel<<<(n_gate + 255) / 256, 256, 0, stream>>>(gate_w, gate_w_bf, n_gate);
    cvt_bf16_kernel<<<(n_memb + 255) / 256, 256, 0, stream>>>(memb, memb_bf, n_memb);

    const size_t lds_bytes = 258048;               // 252 KB of the 320 KB WGP LDS
    tms_fused_kernel<<<Brows / MTILE, 256, lds_bytes, stream>>>(
        cs, ctx, mask, attn_b, gate_b, attn_w_bf, gate_w_bf, memb_bf, out);
}
